// MultiHeadAttention_1743756722773
// MI455X (gfx1250) — matrix operand order repaired and checked by reading, not compiled, not run on hardware
//
#include <hip/hip_runtime.h>

// ---------------- problem constants ----------------
#define B_  4
#define S_  2048
#define E_  1024
#define H_  16
#define D_  64
#define HD_ (H_ * D_)
#define SCALE_ 0.125f   // D^-0.5 = 64^-0.5

typedef __bf16 bf16;
typedef __attribute__((ext_vector_type(16))) __bf16 v16bf;
typedef __attribute__((ext_vector_type(8)))  __bf16 v8bf;
typedef __attribute__((ext_vector_type(8)))  float  v8f;
typedef __attribute__((ext_vector_type(4)))  int    v4i;

union FragU { v16bf v; v8bf h[2]; };

// A-matrix 16x32 bf16 fragment (wave32 layout, ISA 7.12.2):
// lane row = lane&15 ; halves 0..7 = K kb..kb+7 ; halves 8..15 = K kb+16..kb+23 ; kb = (lane&16)?8:0
__device__ __forceinline__ v16bf load_frag_a(const bf16* base, int ld, int lane) {
    int row = lane & 15;
    int kb  = (lane & 16) ? 8 : 0;
    const bf16* p = base + (long)row * ld + kb;
    FragU u;
    u.h[0] = *reinterpret_cast<const v8bf*>(p);
    u.h[1] = *reinterpret_cast<const v8bf*>(p + 16);
    return u.v;
}

// B-matrix 32x16 bf16 fragment from "transposed" storage T[n][k] (k contiguous):
// lane col n = lane&15 ; halves 0..7 = K kb..kb+7 ; halves 8..15 = K kb+16..kb+23 ; kb = (lane&16)?8:0
__device__ __forceinline__ v16bf load_frag_b(const bf16* baseT, int ld, int lane) {
    int n  = lane & 15;
    int kb = (lane & 16) ? 8 : 0;
    const bf16* p = baseT + (long)n * ld + kb;
    FragU u;
    u.h[0] = *reinterpret_cast<const v8bf*>(p);
    u.h[1] = *reinterpret_cast<const v8bf*>(p + 16);
    return u.v;
}

__device__ __forceinline__ v8f wmma_bf16(v16bf a, v16bf b, v8f c) {
    return __builtin_amdgcn_wmma_f32_16x16x32_bf16(false, a, false, b, (short)0, c, false, false);
}

// ---------------- gfx1250 async global->LDS copy (16B per lane) ----------------
// builtin signature (from hipcc diagnostic): (v4i AS1*, v4i AS3*, imm offset, imm cpol)
typedef __attribute__((address_space(1))) v4i* gv4i_p;
typedef __attribute__((address_space(3))) v4i* lv4i_p;

__device__ __forceinline__ void async_copy_b128(const bf16* g, bf16* l) {
#if __has_builtin(__builtin_amdgcn_global_load_async_to_lds_b128)
    __builtin_amdgcn_global_load_async_to_lds_b128((gv4i_p)(void*)g, (lv4i_p)(void*)l, 0, 0);
#else
    *reinterpret_cast<v8bf*>(l) = *reinterpret_cast<const v8bf*>(g);
#endif
}

__device__ __forceinline__ void wait_async0() {
#if __has_builtin(__builtin_amdgcn_global_load_async_to_lds_b128)
#if __has_builtin(__builtin_amdgcn_s_wait_asynccnt)
    __builtin_amdgcn_s_wait_asynccnt(0);
#else
    asm volatile("s_wait_asynccnt 0" ::: "memory");
#endif
#endif
}

// ---------------- kernel 1: fp32 -> bf16 transpose  dst[b][c][r] = src[b][r][c] ----------------
__global__ void k_transpose_bf16(bf16* __restrict__ dst, const float* __restrict__ src,
                                 int rows, int cols, int batch) {
    long rc = (long)rows * cols;
    long total = (long)batch * rc;
    long stride = (long)gridDim.x * blockDim.x;
    for (long i = (long)blockIdx.x * blockDim.x + threadIdx.x; i < total; i += stride) {
        long b   = i / rc;
        long rem = i - b * rc;
        long c   = rem / rows;
        long r   = rem - c * rows;
        dst[i] = (bf16)src[b * rc + r * cols + c];
    }
}

// ---------------- kernel 2: QKV projection ----------------
__global__ __launch_bounds__(256) void k_qkv(const float* __restrict__ x,
                                             const bf16* __restrict__ WqT,
                                             const bf16* __restrict__ WkT,
                                             const bf16* __restrict__ WvT,
                                             bf16* __restrict__ Q,
                                             bf16* __restrict__ K,
                                             bf16* __restrict__ VT) {
    __shared__ bf16 xs[16][E_];                 // 32 KB
    const int tid = threadIdx.x;
    const int blk = blockIdx.x;                 // 0 .. B*S/16-1
    const int b   = blk / (S_ / 16);
    const int s0  = (blk % (S_ / 16)) * 16;

    const float* xrow = x + ((long)b * S_ + s0) * E_;
    for (int i = tid; i < 16 * E_; i += 256) {
        int r = i >> 10, c = i & (E_ - 1);
        xs[r][c] = (bf16)xrow[(long)r * E_ + c];
    }
    __syncthreads();

    const int lane = tid & 31;
    const int w    = tid >> 5;
    const int n    = lane & 15;
    const int hi   = lane >> 4;

    for (int t = w; t < 3 * H_ * 4; t += 8) {
        const int mat = t / (H_ * 4);
        const int rem = t % (H_ * 4);
        const int h   = rem >> 2;
        const int dt  = rem & 3;
        const bf16* Wsel = (mat == 0) ? WqT : ((mat == 1) ? WkT : WvT);
        const bf16* WT = Wsel + ((long)h * D_ + dt * 16) * E_;

        v8f c = {};
        for (int ks = 0; ks < E_; ks += 32) {
            __builtin_prefetch(WT + ks + 256, 0, 1);     // global_prefetch_b8
            v16bf a  = load_frag_a(&xs[0][0] + ks, E_, lane);
            v16bf bb = load_frag_b(WT + ks, E_, lane);
            c = wmma_bf16(a, bb, c);
        }

        const long bh = (long)b * H_ + h;
        if (mat < 2) {
            bf16* out = (mat == 0 ? Q : K) + (bh * S_ + s0) * D_ + dt * 16 + n;
#pragma unroll
            for (int r = 0; r < 8; ++r) {
                int row = r + (hi << 3);
                out[(long)row * D_] = (bf16)c[r];
            }
        } else {
            bf16* out = VT + (bh * D_ + dt * 16 + n) * S_ + s0;
#pragma unroll
            for (int r = 0; r < 8; ++r) {
                int row = r + (hi << 3);
                out[row] = (bf16)c[r];
            }
        }
    }
}

// ---------------- kernel 3: flash attention ----------------
// 8 waves/block, all on the same (b,h); each wave owns one 16-row Q tile.
// K/V 32-column tiles are staged into LDS once per block via async-to-LDS copies,
// double-buffered so the copy for step t+1 overlaps the WMMA work of step t.
#define KPAD 8
#define VPAD 8
__global__ __launch_bounds__(256) void k_attn(const bf16* __restrict__ Q,
                                              const bf16* __restrict__ Km,
                                              const bf16* __restrict__ VT,
                                              bf16* __restrict__ O) {
    __shared__ bf16 Ks[2][32][64 + KPAD];       // ~9 KB
    __shared__ bf16 Vs[2][64][32 + VPAD];       // ~10 KB
    __shared__ bf16 Plds[8][16][32];            // 8 KB
    const int tid  = threadIdx.x;
    const int lane = tid & 31;
    const int w    = tid >> 5;
    const int gw   = blockIdx.x * 8 + w;        // 0 .. B*H*S/16-1
    const int bh   = gw / (S_ / 16);
    const int s0   = (gw % (S_ / 16)) * 16;
    const int b    = bh / H_;
    const int h    = bh % H_;

    // block-uniform head pointers for the cooperative K/V staging
    const int  bh_blk = (blockIdx.x * 8) / (S_ / 16);
    const bf16* KbB = Km + (long)bh_blk * S_ * D_;
    const bf16* VbB = VT + (long)bh_blk * D_ * S_;

    const bf16* Qb = Q + ((long)bh * S_ + s0) * D_;

    const v16bf qa0 = load_frag_a(Qb,      D_, lane);   // d = 0..31
    const v16bf qa1 = load_frag_a(Qb + 32, D_, lane);   // d = 32..63

    float mrow[8], lrow[8];
#pragma unroll
    for (int r = 0; r < 8; ++r) { mrow[r] = -1e30f; lrow[r] = 0.0f; }
    v8f acc[4] = {};                                    // 16 x 64 f32 accumulator

    const int n  = lane & 15;
    const int hi = lane >> 4;

    // cooperative 32-column K/V tile staging: 256 threads x 16B each per tile
    const int kr = tid >> 3, kc = (tid & 7) * 8;        // K tile 32 x 64
    const int vr = tid >> 2, vc = (tid & 3) * 8;        // V tile 64 x 32

    // prologue: stage tile 0
    async_copy_b128(KbB + (long)kr * D_ + kc, &Ks[0][kr][kc]);
    async_copy_b128(VbB + (long)vr * S_ + vc, &Vs[0][vr][vc]);

    for (int t0 = 0; t0 < S_; t0 += 32) {
        const int buf = (t0 >> 5) & 1;
        wait_async0();
        __syncthreads();                                // tile `buf` visible to all waves
        if (t0 + 32 < S_) {                             // overlap next copy with compute
            async_copy_b128(KbB + (long)(t0 + 32 + kr) * D_ + kc, &Ks[buf ^ 1][kr][kc]);
            async_copy_b128(VbB + (long)vr * S_ + (t0 + 32) + vc, &Vs[buf ^ 1][vr][vc]);
        }

        // S = Q * K^T (two 16-col tiles) from LDS
        v8f sl = {}, sh = {};
        sl = wmma_bf16(qa0, load_frag_b(&Ks[buf][0][0],   64 + KPAD, lane), sl);
        sl = wmma_bf16(qa1, load_frag_b(&Ks[buf][0][32],  64 + KPAD, lane), sl);
        sh = wmma_bf16(qa0, load_frag_b(&Ks[buf][16][0],  64 + KPAD, lane), sh);
        sh = wmma_bf16(qa1, load_frag_b(&Ks[buf][16][32], 64 + KPAD, lane), sh);

        // online softmax (rows split across lane halves per the C/D layout)
        float alpha[8];
#pragma unroll
        for (int r = 0; r < 8; ++r) {
            float a0 = sl[r] * SCALE_;
            float a1 = sh[r] * SCALE_;
            float mx = fmaxf(a0, a1);
#pragma unroll
            for (int msk = 1; msk <= 8; msk <<= 1) mx = fmaxf(mx, __shfl_xor(mx, msk, 32));
            float mnew = fmaxf(mrow[r], mx);
            float al   = __expf(mrow[r] - mnew);
            float p0   = __expf(a0 - mnew);
            float p1   = __expf(a1 - mnew);
            sl[r] = p0; sh[r] = p1;
            float rs = p0 + p1;
#pragma unroll
            for (int msk = 1; msk <= 8; msk <<= 1) rs += __shfl_xor(rs, msk, 32);
            lrow[r]  = lrow[r] * al + rs;
            mrow[r]  = mnew;
            alpha[r] = al;
        }
#pragma unroll
        for (int d = 0; d < 4; ++d)
#pragma unroll
            for (int r = 0; r < 8; ++r) acc[d][r] *= alpha[r];

        // C-layout -> A-layout via per-wave LDS staging (bf16)
#pragma unroll
        for (int r = 0; r < 8; ++r) {
            int row = r + (hi << 3);
            Plds[w][row][n]      = (bf16)sl[r];
            Plds[w][row][16 + n] = (bf16)sh[r];
        }
        __builtin_amdgcn_wave_barrier();        // same-wave LDS ordering (DS ops are in-order)
        v16bf pa = load_frag_a(&Plds[w][0][0], 32, lane);

        // O += P * V   (V tile in LDS, transposed: Vs[d][t])
#pragma unroll
        for (int d = 0; d < 4; ++d) {
            v16bf vb = load_frag_b(&Vs[buf][d * 16][0], 32 + VPAD, lane);
            acc[d] = wmma_bf16(pa, vb, acc[d]);
        }
        __builtin_amdgcn_wave_barrier();
    }

    // normalize and emit concat-head layout O[b][s][h*D + d] as bf16
    bf16* Ob = O + ((long)b * S_ + s0) * HD_ + h * D_;
#pragma unroll
    for (int d = 0; d < 4; ++d)
#pragma unroll
        for (int r = 0; r < 8; ++r) {
            int row = r + (hi << 3);
            Ob[(long)row * HD_ + d * 16 + n] = (bf16)(acc[d][r] / lrow[r]);
        }
}

// ---------------- kernel 4: output projection  out = O @ Wo + bo (f32 out) ----------------
__global__ __launch_bounds__(256) void k_proj(const bf16* __restrict__ O,
                                              const bf16* __restrict__ WoT,
                                              const float* __restrict__ bo,
                                              float* __restrict__ out) {
    const int tid  = threadIdx.x;
    const int lane = tid & 31;
    const int w    = tid >> 5;
    const int t    = blockIdx.x * 8 + w;        // 0 .. (BS/16)*(E/16)-1
    const int rowTile = t / (E_ / 16);
    const int colTile = t % (E_ / 16);

    const bf16* Ab = O   + (long)rowTile * 16 * HD_;
    const bf16* Bb = WoT + (long)colTile * 16 * HD_;

    v8f c = {};
    for (int ks = 0; ks < HD_; ks += 32) {
        __builtin_prefetch(Bb + ks + 256, 0, 1);
        v16bf a  = load_frag_a(Ab + ks, HD_, lane);
        v16bf bb = load_frag_b(Bb + ks, HD_, lane);
        c = wmma_bf16(a, bb, c);
    }

    const int n  = lane & 15;
    const int hi = lane >> 4;
    const float bias = bo[colTile * 16 + n];
    float* op = out + (long)rowTile * 16 * E_ + colTile * 16 + n;
#pragma unroll
    for (int r = 0; r < 8; ++r) {
        int row = r + (hi << 3);
        op[(long)row * E_] = c[r] + bias;
    }
}

// ---------------- host-side launch ----------------
extern "C" void kernel_launch(void* const* d_in, const int* in_sizes, int n_in,
                              void* d_out, int out_size, void* d_ws, size_t ws_size,
                              hipStream_t stream) {
    const float* x  = (const float*)d_in[0];
    const float* Wq = (const float*)d_in[1];
    const float* Wk = (const float*)d_in[2];
    const float* Wv = (const float*)d_in[3];
    const float* Wo = (const float*)d_in[4];
    const float* bo = (const float*)d_in[5];
    float* out = (float*)d_out;

    char* ws = (char*)d_ws;
    size_t off = 0;
    auto wsalloc = [&](size_t bytes) -> void* {
        void* p = ws + off;
        off += (bytes + 255) & ~(size_t)255;
        return p;
    };

    bf16* WqT = (bf16*)wsalloc((size_t)H_ * E_ * D_ * sizeof(bf16));   // [H][D][E]
    bf16* WkT = (bf16*)wsalloc((size_t)H_ * E_ * D_ * sizeof(bf16));
    bf16* WvT = (bf16*)wsalloc((size_t)H_ * E_ * D_ * sizeof(bf16));
    bf16* WoT = (bf16*)wsalloc((size_t)HD_ * E_    * sizeof(bf16));    // [E][HD]
    bf16* Qd  = (bf16*)wsalloc((size_t)B_ * H_ * S_ * D_ * sizeof(bf16)); // [BH][S][D]
    bf16* Kd  = (bf16*)wsalloc((size_t)B_ * H_ * S_ * D_ * sizeof(bf16)); // [BH][S][D]
    bf16* VTd = (bf16*)wsalloc((size_t)B_ * H_ * D_ * S_ * sizeof(bf16)); // [BH][D][S]
    bf16* Od  = (bf16*)wsalloc((size_t)B_ * S_ * HD_     * sizeof(bf16)); // [BS][HD]
    (void)ws_size; (void)in_sizes; (void)n_in; (void)out_size;

    // 1) weight convert+transpose (bf16)
    k_transpose_bf16<<<1024, 256, 0, stream>>>(WqT, Wq, E_, D_, H_);
    k_transpose_bf16<<<1024, 256, 0, stream>>>(WkT, Wk, E_, D_, H_);
    k_transpose_bf16<<<1024, 256, 0, stream>>>(WvT, Wv, E_, D_, H_);
    k_transpose_bf16<<<1024, 256, 0, stream>>>(WoT, Wo, HD_, E_, 1);

    // 2) QKV projections (WMMA)
    k_qkv<<<B_ * S_ / 16, 256, 0, stream>>>(x, WqT, WkT, WvT, Qd, Kd, VTd);

    // 3) flash attention (WMMA + async-to-LDS double buffering)
    k_attn<<<(B_ * H_ * S_ / 16) / 8, 256, 0, stream>>>(Qd, Kd, VTd, Od);

    // 4) output projection + bias (WMMA)
    k_proj<<<((B_ * S_ / 16) * (E_ / 16)) / 8, 256, 0, stream>>>(Od, WoT, bo, out);
}
